// Margin_71614284694204
// MI455X (gfx1250) — matrix operand order repaired and checked by reading, not compiled, not run on hardware
//
#include <hip/hip_runtime.h>

typedef __attribute__((ext_vector_type(16))) _Float16 v16h;
typedef __attribute__((ext_vector_type(8)))  _Float16 v8h;
typedef __attribute__((ext_vector_type(8)))  float    v8f;

static constexpr int kN = 1024;    // batch
static constexpr int kD = 512;     // feature dim (K)
static constexpr int kC = 65536;   // classes (N of GEMM)

#define MARGIN_V 0.1f
#define SCALE_V  32.0f

// Deterministic hash -> approx N(0,1) margin noise (stands in for jax key-42
// threefry draw; only label-matching entries are consumed).
__device__ __forceinline__ float margin_noise(unsigned n, unsigned c)
{
    unsigned s = n * 65599u + c * 2654435769u + 42u;
    s ^= s >> 16; s *= 2246822519u; s ^= s >> 13; s *= 3266489917u; s ^= s >> 16;
    unsigned s2 = s * 747796405u + 2891336453u;
    s2 ^= s2 >> 17; s2 *= 0xed5ad4bbu; s2 ^= s2 >> 11;
    const float u1 = (float)(s >> 8) * (1.0f / 16777216.0f) + 1e-7f;
    const float u2 = (float)(s2 >> 8) * (1.0f / 16777216.0f);
    const float r  = sqrtf(-2.0f * __logf(u1));
    const float z  = r * __cosf(6.28318530718f * u2);
    return MARGIN_V + 0.1f * MARGIN_V * z;
}

// ---------------------------------------------------------------------------
// Kernel 1: L2-normalize rows of feature[1024x512] and W[65536x512] (fp32)
// into f16 workspace rows; also precompute the per-row margin scalar
// mr[n] = margin_noise(n, labels[n]).  One wave32 per row.
// ---------------------------------------------------------------------------
__global__ __launch_bounds__(256)
void norm_rows_f16(const float* __restrict__ feat,
                   const float* __restrict__ W,
                   const int*   __restrict__ labels,
                   _Float16* __restrict__ xh,
                   _Float16* __restrict__ wh,
                   float*    __restrict__ mr)
{
    const int wave = (blockIdx.x * blockDim.x + threadIdx.x) >> 5;
    const int lane = threadIdx.x & 31;
    const int totalRows = kN + kC;
    if (wave >= totalRows) return;

    const float* src;
    _Float16*    dst;
    if (wave < kN) {
        src = feat + (size_t)wave * kD;
        dst = xh   + (size_t)wave * kD;
    } else {
        const int r = wave - kN;
        src = W  + (size_t)r * kD;
        dst = wh + (size_t)r * kD;
    }

    const float4* s4 = (const float4*)src + lane * 4;
    float4 f[4];
    f[0] = s4[0]; f[1] = s4[1]; f[2] = s4[2]; f[3] = s4[3];

    float ss = 0.0f;
#pragma unroll
    for (int i = 0; i < 4; ++i)
        ss += f[i].x * f[i].x + f[i].y * f[i].y + f[i].z * f[i].z + f[i].w * f[i].w;

#pragma unroll
    for (int m = 16; m >= 1; m >>= 1)
        ss += __shfl_xor(ss, m, 32);

    const float rn = rsqrtf(fmaxf(ss, 1e-12f));

    _Float16 h[16];
#pragma unroll
    for (int i = 0; i < 4; ++i) {
        h[4 * i + 0] = (_Float16)(f[i].x * rn);
        h[4 * i + 1] = (_Float16)(f[i].y * rn);
        h[4 * i + 2] = (_Float16)(f[i].z * rn);
        h[4 * i + 3] = (_Float16)(f[i].w * rn);
    }
    uint4* d4 = (uint4*)(dst + lane * 16);
    d4[0] = *(const uint4*)&h[0];
    d4[1] = *(const uint4*)&h[8];

    // per-row margin scalar for feature rows (one lane does it)
    if (wave < kN && lane == 0)
        mr[wave] = margin_noise((unsigned)wave, (unsigned)labels[wave]);
}

// --- CDNA5 async global->LDS copy (ASYNCcnt-tracked, no VGPR round trip) ---
__device__ __forceinline__ void async_copy_b128(unsigned lds_off, const _Float16* g)
{
    asm volatile("global_load_async_to_lds_b128 %0, %1, off"
                 :
                 : "v"(lds_off), "v"(g)
                 : "memory");
}
__device__ __forceinline__ void wait_async0()
{
    asm volatile("s_wait_asynccnt 0x0" ::: "memory");
}

// ---------------------------------------------------------------------------
// Kernel 2: f16 WMMA GEMM  s = xh @ whT  with fused margin-logits epilogue.
// Block tile 128(M) x 256(N); 8 wave32 as 2x4; each wave 4x4 WMMA tiles.
// Double-buffered LDS fed by global_load_async_to_lds_b128.
// ---------------------------------------------------------------------------
#define BM 128
#define BN 256
#define BK 32
#define LDA 40   // BK + 8 halfs pad (16B) against LDS bank conflicts
#define LDB 40

__global__ __launch_bounds__(256)
void margin_gemm(const _Float16* __restrict__ xh,
                 const _Float16* __restrict__ wh,
                 const int*   __restrict__ labels,
                 const float* __restrict__ mr,
                 float* __restrict__ out)
{
    __shared__ _Float16 As[2][BM * LDA];   // 2 x 10240 B
    __shared__ _Float16 Bs[2][BN * LDB];   // 2 x 20480 B
    __shared__ int      labs[BM];          // 512 B
    __shared__ float    mrs[BM];           // 512 B

    const int tid  = threadIdx.x;
    const int lane = tid & 31;
    const int wave = tid >> 5;      // 0..7
    const int wm   = wave >> 2;     // 0..1  (M)
    const int wn   = wave & 3;      // 0..3  (N)

    const int blockM = blockIdx.y * BM;
    const int blockN = blockIdx.x * BN;

    const int lhalf = lane & 15;
    const int lsel  = lane >> 4;

    if (tid < BM) {
        labs[tid] = labels[blockM + tid];
        mrs[tid]  = mr[blockM + tid];
    }

    const unsigned asOff0 = (unsigned)(uintptr_t)&As[0][0];
    const unsigned asOff1 = (unsigned)(uintptr_t)&As[1][0];
    const unsigned bsOff0 = (unsigned)(uintptr_t)&Bs[0][0];
    const unsigned bsOff1 = (unsigned)(uintptr_t)&Bs[1][0];

    // issue this thread's share of one K-step tile as async b128 copies
    auto stage = [&](int k0, int buf) {
        const unsigned aBase = buf ? asOff1 : asOff0;
        const unsigned bBase = buf ? bsOff1 : bsOff0;
        // A: 128 rows x 64B -> 512 x b128, 2 per thread
#pragma unroll
        for (int i = 0; i < 2; ++i) {
            const int idx = tid + i * 256;
            const int row = idx >> 2;
            const int seg = (idx & 3) * 8;
            async_copy_b128(aBase + (unsigned)(row * LDA + seg) * 2u,
                            xh + (size_t)(blockM + row) * kD + k0 + seg);
        }
        // B: 256 rows x 64B -> 1024 x b128, 4 per thread
#pragma unroll
        for (int i = 0; i < 4; ++i) {
            const int idx = tid + i * 256;
            const int row = idx >> 2;
            const int seg = (idx & 3) * 8;
            async_copy_b128(bBase + (unsigned)(row * LDB + seg) * 2u,
                            wh + (size_t)(blockN + row) * kD + k0 + seg);
        }
    };

    v8f acc[4][4];
#pragma unroll
    for (int i = 0; i < 4; ++i)
#pragma unroll
        for (int j = 0; j < 4; ++j)
            acc[i][j] = (v8f){};

    stage(0, 0);

    int buf = 0;
    for (int k0 = 0; k0 < kD; k0 += BK, buf ^= 1) {
        wait_async0();        // my async copies into buf are done
        __syncthreads();      // everyone's done; prior buf reads finished too
        if (k0 + BK < kD) stage(k0 + BK, buf ^ 1);

        const _Float16* __restrict__ A = buf ? &As[1][0] : &As[0][0];
        const _Float16* __restrict__ B = buf ? &Bs[1][0] : &Bs[0][0];

        // A 16x32 frags: lanes 0-15 row m K{0..7,16..23}; lanes 16-31 K{8..15,24..31}
        v16h aF[4];
#pragma unroll
        for (int tm = 0; tm < 4; ++tm) {
            const int row = wm * 64 + tm * 16 + lhalf;
            const v8h lo = *(const v8h*)(A + row * LDA + lsel * 8);
            const v8h hi = *(const v8h*)(A + row * LDA + 16 + lsel * 8);
            aF[tm] = __builtin_shufflevector(lo, hi,
                0, 1, 2, 3, 4, 5, 6, 7, 8, 9, 10, 11, 12, 13, 14, 15);
        }
        // B 32x16 frags: lanes 0-15 col n K{0..7,16..23}; lanes 16-31 K{8..15,24..31}
        v16h bF[4];
#pragma unroll
        for (int tn = 0; tn < 4; ++tn) {
            const int col = wn * 64 + tn * 16 + lhalf;
            const v8h lo = *(const v8h*)(B + col * LDB + lsel * 8);
            const v8h hi = *(const v8h*)(B + col * LDB + 16 + lsel * 8);
            bF[tn] = __builtin_shufflevector(lo, hi,
                0, 1, 2, 3, 4, 5, 6, 7, 8, 9, 10, 11, 12, 13, 14, 15);
        }

#pragma unroll
        for (int tm = 0; tm < 4; ++tm)
#pragma unroll
            for (int tn = 0; tn < 4; ++tn)
                acc[tm][tn] = __builtin_amdgcn_wmma_f32_16x16x32_f16(
                    false, aF[tm], false, bF[tn],
                    (short)0, acc[tm][tn], false, false);
    }

    // ---- epilogue: dist = 2 - 2s (+ per-row margin on label hit);
    //      out = 32 * rcp(dist).  Branchless; stores use immediate offsets.
    // C/D layout: VGPR r, lanes 0-15 -> (M=r, N=lane); lanes 16-31 -> (M=r+8, N=lane-16)
#pragma unroll
    for (int tm = 0; tm < 4; ++tm) {
        const int rbase = wm * 64 + tm * 16 + lsel * 8;   // local row of r=0
        int   l[8];
        float m[8];
        *(int4*)&l[0]   = *(const int4*)&labs[rbase];
        *(int4*)&l[4]   = *(const int4*)&labs[rbase + 4];
        *(float4*)&m[0] = *(const float4*)&mrs[rbase];
        *(float4*)&m[4] = *(const float4*)&mrs[rbase + 4];
#pragma unroll
        for (int tn = 0; tn < 4; ++tn) {
            const int gc = blockN + wn * 64 + tn * 16 + lhalf;
            float* op = out + ((size_t)(blockM + rbase) << 16) + gc;
#pragma unroll
            for (int r = 0; r < 8; ++r) {
                const float s = acc[tm][tn][r];
                float dist = fmaf(-2.0f, s, 2.0f);
                dist += (l[r] == gc) ? m[r] : 0.0f;
                op[(size_t)r << 16] = SCALE_V * __builtin_amdgcn_rcpf(dist);
            }
        }
    }
}

// ---------------------------------------------------------------------------
extern "C" void kernel_launch(void* const* d_in, const int* in_sizes, int n_in,
                              void* d_out, int out_size, void* d_ws, size_t ws_size,
                              hipStream_t stream)
{
    const float* feat   = (const float*)d_in[0];
    const int*   labels = (const int*)d_in[1];
    const float* W      = (const float*)d_in[2];
    float*       out    = (float*)d_out;

    _Float16* xh = (_Float16*)d_ws;                 // 1 MB
    _Float16* wh = xh + (size_t)kN * kD;            // 64 MB
    float*    mr = (float*)(wh + (size_t)kC * kD);  // 4 KB

    const int totalRows = kN + kC;                  // 66560 rows, 1 wave each
    const int blocks = (totalRows + 7) / 8;         // 8 waves per block
    norm_rows_f16<<<blocks, 256, 0, stream>>>(feat, W, labels, xh, wh, mr);

    dim3 grid(kC / BN, kN / BM);                    // 256 x 8
    margin_gemm<<<grid, 256, 0, stream>>>(xh, wh, labels, mr, out);
}
